// HardMoE_82016695484511
// MI455X (gfx1250) — matrix operand order repaired and checked by reading, not compiled, not run on hardware
//
#include <hip/hip_runtime.h>
#include <hip/hip_bf16.h>

#define IN_DIM  1024
#define HID_DIM 4096
#define OUT_DIM 1024
#define N_EXP   8
#define BATCH   8192
#define PAD_ROWS (BATCH + N_EXP * 128)   // 9216: per-expert segments padded to 128

typedef __attribute__((ext_vector_type(16))) __bf16 v16bf;
typedef __attribute__((ext_vector_type(8)))  float  v8f;

#if __has_builtin(__builtin_amdgcn_global_load_async_to_lds_b128)
#define HAVE_ASYNC_LDS 1
#endif

// builtin signature: void(v4i addrspace(1)* src, v4i addrspace(3)* dst, imm offset, imm cpol)
typedef int v4i __attribute__((vector_size(16)));
typedef __attribute__((address_space(1))) v4i gv4i;
typedef __attribute__((address_space(3))) v4i lv4i;

__device__ __forceinline__ gv4i* as_global(const void* p) {
    return (gv4i*)(unsigned long long)p;                  // flat VA == global VA
}
__device__ __forceinline__ lv4i* as_lds(void* p) {
    return (lv4i*)(unsigned int)(unsigned long long)p;    // low 32 bits = LDS offset
}
__device__ __forceinline__ void wait_async0() {
#if __has_builtin(__builtin_amdgcn_s_wait_asynccnt)
    __builtin_amdgcn_s_wait_asynccnt(0);
#else
    asm volatile("s_wait_asynccnt 0x0" ::: "memory");
#endif
}

__device__ __forceinline__ unsigned short f2bf(float f) {
    unsigned int u = __float_as_uint(f);
    u += 0x7FFFu + ((u >> 16) & 1u);     // round-to-nearest-even
    return (unsigned short)(u >> 16);
}

// ---------------- f32 -> bf16 conversion (vectorized x4) ----------------
__global__ void cvt_kernel(const float* __restrict__ in,
                           unsigned short* __restrict__ out, int n4) {
    int i = blockIdx.x * blockDim.x + threadIdx.x;
    if (i >= n4) return;
    float4 f = ((const float4*)in)[i];
    unsigned int lo = (unsigned int)f2bf(f.x) | ((unsigned int)f2bf(f.y) << 16);
    unsigned int hi = (unsigned int)f2bf(f.z) | ((unsigned int)f2bf(f.w) << 16);
    ((uint2*)out)[i] = make_uint2(lo, hi);
}

// ------- f32 [K][N] -> bf16 transposed [N][K], LDS-tiled (32k x 64n tiles) -------
__global__ __launch_bounds__(256) void cvtT_kernel(const float* __restrict__ in,
                                                   unsigned short* __restrict__ out,
                                                   int K, int N) {
    __shared__ __align__(16) unsigned short t[64 * 40];   // [n][k], padded stride 40
    const size_t mat = (size_t)blockIdx.z * K * N;
    const float* inp = in + mat;
    unsigned short* outp = out + mat;
    const int k0 = blockIdx.x * 32;
    const int n0 = blockIdx.y * 64;
    const int tid = threadIdx.x;
    #pragma unroll
    for (int q = 0; q < 2; ++q) {
        int idx = tid * 2 + q;            // float4 index within tile (512 total)
        int kk  = idx >> 4;               // 16 float4 per k-row
        int nn  = (idx & 15) * 4;
        float4 f = *(const float4*)(inp + (size_t)(k0 + kk) * N + n0 + nn);
        t[(nn + 0) * 40 + kk] = f2bf(f.x);
        t[(nn + 1) * 40 + kk] = f2bf(f.y);
        t[(nn + 2) * 40 + kk] = f2bf(f.z);
        t[(nn + 3) * 40 + kk] = f2bf(f.w);
    }
    __syncthreads();
    const int nn = tid >> 2;
    const int ks = (tid & 3) * 8;
    uint4 v = *(const uint4*)(t + nn * 40 + ks);
    *(uint4*)(outp + (size_t)(n0 + nn) * K + k0 + ks) = v;
}

// ---------------- router: one wave32 per token ----------------
__global__ __launch_bounds__(256) void router_kernel(
    const float* __restrict__ x, const float* __restrict__ rw,
    const float* __restrict__ rbias, int* __restrict__ idx,
    int* __restrict__ counts, float* __restrict__ rout) {
    int wave = threadIdx.x >> 5;
    int lane = threadIdx.x & 31;
    int t = blockIdx.x * 8 + wave;
    if (t >= BATCH) return;
    const float* xr = x + (size_t)t * IN_DIM;
    float acc[8] = {0.f,0.f,0.f,0.f,0.f,0.f,0.f,0.f};
    for (int it = 0; it < IN_DIM / 32; ++it) {
        int k = lane + it * 32;
        float xv = xr[k];
        const float4* w4 = (const float4*)(rw + (size_t)k * N_EXP);
        float4 w0 = w4[0], w1 = w4[1];
        acc[0] += xv * w0.x; acc[1] += xv * w0.y;
        acc[2] += xv * w0.z; acc[3] += xv * w0.w;
        acc[4] += xv * w1.x; acc[5] += xv * w1.y;
        acc[6] += xv * w1.z; acc[7] += xv * w1.w;
    }
    for (int off = 16; off > 0; off >>= 1) {
        #pragma unroll
        for (int e = 0; e < 8; ++e)
            acc[e] += __shfl_down(acc[e], off, 32);
    }
    if (lane == 0) {
        int best = 0; float bv = acc[0] + rbias[0];
        #pragma unroll
        for (int e = 1; e < 8; ++e) {
            float v = acc[e] + rbias[e];
            if (v > bv) { bv = v; best = e; }   // strict > keeps first max (jnp.argmax)
        }
        idx[t] = best;
        atomicAdd(&counts[best], 1);
        float* rr = rout + (size_t)t * N_EXP;
        #pragma unroll
        for (int e = 0; e < 8; ++e) rr[e] = (e == best) ? 1.0f : 0.0f;
    }
}

// ---------------- routing bookkeeping ----------------
__global__ void init_kernel(int* perm, int* counts, int* cursors) {
    int t = blockIdx.x * blockDim.x + threadIdx.x;
    if (t < PAD_ROWS) perm[t] = -1;
    if (t < N_EXP) { counts[t] = 0; cursors[t] = 0; }
}

__global__ void prefix_kernel(const int* counts, int* seg) {
    if (threadIdx.x == 0 && blockIdx.x == 0) {
        int s = 0;
        for (int e = 0; e < N_EXP; ++e) {
            seg[e] = s;
            s += (counts[e] + 127) & ~127;   // 128-align: tiles never straddle experts
        }
        seg[N_EXP] = s;
    }
}

__global__ void scatter_kernel(const int* idx, const int* seg, int* cursors, int* perm) {
    int t = blockIdx.x * blockDim.x + threadIdx.x;
    if (t >= BATCH) return;
    int e = idx[t];
    int p = atomicAdd(&cursors[e], 1);
    perm[seg[e] + p] = t;
}

// ---------------- 128x128x32 bf16 WMMA GEMM, double-buffered LDS ----------------
// B is pre-transposed: Bt[N][K] bf16 (per-expert blocks if GATHER).
// Staging uses GLOBAL_LOAD_ASYNC_TO_LDS_B128 (ASYNCcnt) when available.
template <bool GATHER>
__global__ __launch_bounds__(256) void moe_gemm(
    const unsigned short* __restrict__ A,    // bf16, lda == K
    const unsigned short* __restrict__ Bt,   // bf16, [N][K]
    const float* __restrict__ bias,
    unsigned short* __restrict__ outH,
    float* __restrict__ outF,
    const int* __restrict__ perm,
    const int* __restrict__ seg,
    int K, int N) {
    const int tid = threadIdx.x;
    const int rb  = blockIdx.x * 128;
    const int c0  = blockIdx.y * 128;

    const unsigned short* Bm = Bt;
    const float* biasp = bias;
    if (GATHER) {
        if (rb >= seg[N_EXP]) return;        // uniform per block: safe early exit
        int e = 0;
        #pragma unroll
        for (int q = 1; q < N_EXP; ++q) if (rb >= seg[q]) e = q;
        Bm    = Bt   + (size_t)e * IN_DIM * HID_DIM;
        biasp = bias + (size_t)e * HID_DIM;
    }

    __shared__ __align__(16) unsigned short As[2][128 * 40];  // [row][k], stride 40
    __shared__ __align__(16) unsigned short Bs[2][128 * 40];  // [col][k], stride 40

    const int lane  = tid & 31;
    const int wid   = tid >> 5;
    const int wr    = wid & 3;     // wave row group: 32 rows each
    const int wc    = wid >> 2;    // wave col group: 64 cols each
    const int lcol  = lane & 15;
    const int lhalf = lane >> 4;

    v8f acc[2][4] = {};

    // cooperative staging: each thread owns 16 contiguous bf16 of one A row and one B row
    const int srow = tid >> 1;            // 0..127
    const int sseg = (tid & 1) * 16;      // 0/16
    int atok = GATHER ? perm[rb + srow] : (rb + srow);
    if (atok < 0) atok = 0;               // padding rows: load real data, never stored
    const unsigned short* aptr = A  + (size_t)atok * K + sseg;
    const unsigned short* bptr = Bm + (size_t)(c0 + srow) * K + sseg;
    const int soff = srow * 40 + sseg;

    const int nk = K / 32;
    int cur = 0;

    // ---- prologue: stage tile 0 into buffer 0 ----
#ifdef HAVE_ASYNC_LDS
    {
        __builtin_amdgcn_global_load_async_to_lds_b128(as_global(aptr), as_lds(As[0] + soff), 0, 0);
        __builtin_amdgcn_global_load_async_to_lds_b128(as_global(aptr), as_lds(As[0] + soff), 16, 0);
        __builtin_amdgcn_global_load_async_to_lds_b128(as_global(bptr), as_lds(Bs[0] + soff), 0, 0);
        __builtin_amdgcn_global_load_async_to_lds_b128(as_global(bptr), as_lds(Bs[0] + soff), 16, 0);
    }
#else
    {
        uint4 a0 = ((const uint4*)aptr)[0], a1 = ((const uint4*)aptr)[1];
        uint4 b0 = ((const uint4*)bptr)[0], b1 = ((const uint4*)bptr)[1];
        uint4* da = (uint4*)(As[0] + soff);
        uint4* db = (uint4*)(Bs[0] + soff);
        da[0] = a0; da[1] = a1;
        db[0] = b0; db[1] = b1;
    }
#endif

    for (int it = 0; it < nk; ++it) {
#ifdef HAVE_ASYNC_LDS
        wait_async0();                    // my async stores to buf[cur] have landed
#endif
        __syncthreads();                  // buf[cur] ready; buf[cur^1] free
        const bool more = (it + 1 < nk);

#ifdef HAVE_ASYNC_LDS
        if (more) {                       // async-stage next tile straight into LDS
            const int ko = (it + 1) * 32;
            unsigned short* an = As[cur ^ 1] + soff;
            unsigned short* bn = Bs[cur ^ 1] + soff;
            __builtin_amdgcn_global_load_async_to_lds_b128(as_global(aptr + ko), as_lds(an), 0, 0);
            __builtin_amdgcn_global_load_async_to_lds_b128(as_global(aptr + ko), as_lds(an), 16, 0);
            __builtin_amdgcn_global_load_async_to_lds_b128(as_global(bptr + ko), as_lds(bn), 0, 0);
            __builtin_amdgcn_global_load_async_to_lds_b128(as_global(bptr + ko), as_lds(bn), 16, 0);
        }
#else
        uint4 na0 = {0,0,0,0}, na1 = {0,0,0,0}, nb0 = {0,0,0,0}, nb1 = {0,0,0,0};
        if (more) {
            const int ko = (it + 1) * 32;
            na0 = *(const uint4*)(aptr + ko); na1 = *(const uint4*)(aptr + ko + 8);
            nb0 = *(const uint4*)(bptr + ko); nb1 = *(const uint4*)(bptr + ko + 8);
        }
#endif

        // ---- fragments per documented 16x16x32 bf16 VGPR layouts ----
        v16bf af[2], bfg[4];
        #pragma unroll
        for (int i = 0; i < 2; ++i) {
            // lanes 0-15: K 0..7 then 16..23 ; lanes 16-31: K 8..15 then 24..31
            const unsigned short* base = As[cur] + (wr * 32 + i * 16 + lcol) * 40 + lhalf * 8;
            union { v16bf v; uint4 q[2]; } u;
            u.q[0] = *(const uint4*)(base);
            u.q[1] = *(const uint4*)(base + 16);
            af[i] = u.v;
        }
        #pragma unroll
        for (int j = 0; j < 4; ++j) {
            // lanes 0-15: K 0..7 then 16..23 ; lanes 16-31: K 8..15 then 24..31
            const unsigned short* base = Bs[cur] + (wc * 64 + j * 16 + lcol) * 40 + lhalf * 8;
            union { v16bf v; uint4 q[2]; } u;
            u.q[0] = *(const uint4*)(base);
            u.q[1] = *(const uint4*)(base + 16);
            bfg[j] = u.v;
        }

        #pragma unroll
        for (int i = 0; i < 2; ++i)
            #pragma unroll
            for (int j = 0; j < 4; ++j)
                acc[i][j] = __builtin_amdgcn_wmma_f32_16x16x32_bf16(
                    false, af[i], false, bfg[j], (short)0, acc[i][j], false, false);

#ifndef HAVE_ASYNC_LDS
        if (more) {
            uint4* da = (uint4*)(As[cur ^ 1] + soff);
            uint4* db = (uint4*)(Bs[cur ^ 1] + soff);
            da[0] = na0; da[1] = na1;
            db[0] = nb0; db[1] = nb1;
        }
#endif
        if (more) cur ^= 1;
    }

    // ---- epilogue: C/D layout: VGPR v, lanes 0-15 -> M=v, lanes 16-31 -> M=v+8 ----
    #pragma unroll
    for (int i = 0; i < 2; ++i) {
        #pragma unroll
        for (int v = 0; v < 8; ++v) {
            int row_l = wr * 32 + i * 16 + v + 8 * lhalf;
            int tok = GATHER ? perm[rb + row_l] : (rb + row_l);
            if (GATHER && tok < 0) continue;       // padding row
            #pragma unroll
            for (int j = 0; j < 4; ++j) {
                int gcol = c0 + wc * 64 + j * 16 + lcol;
                float val = acc[i][j][v] + biasp[gcol];
                if (GATHER) outH[(size_t)tok * N + gcol] = f2bf(val);
                else        outF[(size_t)tok * N + gcol] = val;
            }
        }
    }
}

// ---------------- launch ----------------
extern "C" void kernel_launch(void* const* d_in, const int* in_sizes, int n_in,
                              void* d_out, int out_size, void* d_ws, size_t ws_size,
                              hipStream_t stream) {
    (void)in_sizes; (void)n_in; (void)out_size; (void)ws_size;
    const float* x  = (const float*)d_in[0];
    const float* rw = (const float*)d_in[1];
    const float* rb = (const float*)d_in[2];
    const float* ew = (const float*)d_in[3];
    const float* eb = (const float*)d_in[4];
    const float* pw = (const float*)d_in[5];
    const float* pb = (const float*)d_in[6];

    float* out_main = (float*)d_out;                       // [BATCH, OUT_DIM]
    float* out_rout = out_main + (size_t)BATCH * OUT_DIM;  // [BATCH, N_EXP]

    char* ws = (char*)d_ws;
    size_t off = 0;
    auto alloc = [&](size_t bytes) -> char* {
        char* p = ws + off;
        off = (off + bytes + 255) & ~(size_t)255;
        return p;
    };
    unsigned short* x_bf  = (unsigned short*)alloc((size_t)BATCH * IN_DIM * 2);
    unsigned short* ew_t  = (unsigned short*)alloc((size_t)N_EXP * IN_DIM * HID_DIM * 2); // [E][HID][IN]
    unsigned short* pw_t  = (unsigned short*)alloc((size_t)HID_DIM * OUT_DIM * 2);        // [OUT][HID]
    unsigned short* h_bf  = (unsigned short*)alloc((size_t)BATCH * HID_DIM * 2);
    int* idx     = (int*)alloc((size_t)BATCH * 4);
    int* perm    = (int*)alloc((size_t)PAD_ROWS * 4);
    int* counts  = (int*)alloc((size_t)N_EXP * 4);
    int* cursors = (int*)alloc((size_t)N_EXP * 4);
    int* seg     = (int*)alloc((size_t)(N_EXP + 1) * 4);

    int n4 = BATCH * IN_DIM / 4;
    cvt_kernel<<<(n4 + 255) / 256, 256, 0, stream>>>(x, x_bf, n4);
    // expert_w [E][IN][HID] -> ew_t [E][HID][IN]
    cvtT_kernel<<<dim3(IN_DIM / 32, HID_DIM / 64, N_EXP), 256, 0, stream>>>(ew, ew_t, IN_DIM, HID_DIM);
    // proj_w [HID][OUT] -> pw_t [OUT][HID]
    cvtT_kernel<<<dim3(HID_DIM / 32, OUT_DIM / 64, 1), 256, 0, stream>>>(pw, pw_t, HID_DIM, OUT_DIM);

    init_kernel<<<(PAD_ROWS + 255) / 256, 256, 0, stream>>>(perm, counts, cursors);
    router_kernel<<<BATCH / 8, 256, 0, stream>>>(x, rw, rb, idx, counts, out_rout);
    prefix_kernel<<<1, 32, 0, stream>>>(counts, seg);
    scatter_kernel<<<(BATCH + 255) / 256, 256, 0, stream>>>(idx, seg, cursors, perm);

    dim3 g1(PAD_ROWS / 128, HID_DIM / 128);   // 72 x 32
    moe_gemm<true><<<g1, 256, 0, stream>>>(x_bf, ew_t, eb, h_bf, nullptr,
                                           perm, seg, IN_DIM, HID_DIM);
    dim3 g2(BATCH / 128, OUT_DIM / 128);      // 64 x 8
    moe_gemm<false><<<g2, 256, 0, stream>>>(h_bf, pw_t, pb, nullptr, out_main,
                                            nullptr, nullptr, HID_DIM, OUT_DIM);
}
